// MultiheadAttention_71073118814720
// MI455X (gfx1250) — matrix operand order repaired and checked by reading, not compiled, not run on hardware
//
#include <hip/hip_runtime.h>

#define B_  2
#define LQ_ 2048
#define LK_ 2048
#define D_  1024
#define H_  16
#define E_  64
#define HE_ (H_ * E_)
#define TAU 0.125f   // E^-0.5

typedef __attribute__((ext_vector_type(16))) __bf16 v16bf;
typedef __attribute__((ext_vector_type(8)))  __bf16 v8bf;
typedef __attribute__((ext_vector_type(4)))  __bf16 v4bf;
typedef __attribute__((ext_vector_type(8)))  float  v8f;

static __device__ __forceinline__ v8f wmma_bf16(v16bf a, v16bf b, v8f c) {
  // 8 args: (neg_a, A, neg_b, B, c_mod, C, reuse_a, reuse_b)
  return __builtin_amdgcn_wmma_f32_16x16x32_bf16(false, a, false, b, (short)0, c,
                                                 false, false);
}

// A-operand (16x32 bf16). Lane l holds row (l&15);
// lanes 0-15: K = kofs+0..15; lanes 16-31: K = kofs+16..31 (contiguous, matches B).
static __device__ __forceinline__ v16bf load_a16(const __bf16* base, int ld, int kofs) {
  const int lane = threadIdx.x & 31;
  const __bf16* p = base + (size_t)(lane & 15) * ld + kofs + ((lane >> 4) << 4);
  return *(const v16bf*)p;
}

// B-operand (32x16 bf16). Lane l supplies column (l&15);
// lanes 0-15 hold K = kofs+0..15, lanes 16-31 hold K = kofs+16..31 (contiguous,
// 32B-aligned at every call site) -> single v16bf load, one contiguous tuple.
static __device__ __forceinline__ v16bf load_b16(const __bf16* base, int ld, int kofs) {
  const int lane = threadIdx.x & 31;
  const __bf16* p = base + (size_t)(lane & 15) * ld + kofs + ((lane >> 4) << 4);
  return *(const v16bf*)p;
}

// ---------------- f32 -> bf16 convert ----------------
__global__ __launch_bounds__(256) void cvt_f32_bf16(const float* __restrict__ in,
                                                    __bf16* __restrict__ out, int n) {
  int i = (blockIdx.x * 256 + threadIdx.x) * 4;
  if (i + 3 < n) {
    float4 f = *(const float4*)(in + i);
    v4bf o;
    o[0] = (__bf16)f.x; o[1] = (__bf16)f.y; o[2] = (__bf16)f.z; o[3] = (__bf16)f.w;
    *(v4bf*)(out + i) = o;
  }
}

// ---------------- per-head input projections ----------------
// z=0: qh[bh,Lq,E] = q x Wq^T + bq     z=1: kh[bh,Lk,E]
// z=2: vhT[bh,E,Lk] (transposed store for the ctx GEMM B-operand)
__global__ __launch_bounds__(256)
__attribute__((amdgpu_waves_per_eu(2))) void proj_kernel(
    const __bf16* __restrict__ qb, const __bf16* __restrict__ kb,
    const __bf16* __restrict__ vb,
    const __bf16* __restrict__ Wq, const __bf16* __restrict__ Wk,
    const __bf16* __restrict__ Wv,
    const float* __restrict__ bq, const float* __restrict__ bk,
    const float* __restrict__ bv,
    __bf16* __restrict__ qh, __bf16* __restrict__ kh, __bf16* __restrict__ vhT) {
  const int which = blockIdx.z;
  const int bh = blockIdx.x, b = bh / H_, h = bh % H_;
  const int wave = threadIdx.x >> 5, lane = threadIdx.x & 31;
  const int row0 = blockIdx.y * 128 + wave * 16;

  const __bf16* X = (which == 0) ? qb : (which == 1) ? kb : vb;
  const __bf16* W = (which == 0) ? Wq : (which == 1) ? Wk : Wv;
  const float* bias = (which == 0) ? bq : (which == 1) ? bk : bv;

  const __bf16* Xb = X + ((size_t)b * LQ_ + row0) * D_;
  const __bf16* Wh = W + (size_t)h * E_ * D_;

  v8f acc[4] = {};
  // All tiles for both K-steps load at the top of the body: no loop-carried
  // tile registers (no phis -> no copies). waves_per_eu(2) widens the VGPR
  // budget so the scheduler keeps each tile in its own tuple instead of
  // serializing through one, letting the load clause run ahead of the WMMAs.
#pragma unroll 1
  for (int kk = 0; kk < D_; kk += 64) {
    v16bf a0 = load_a16(Xb, D_, kk);
    v16bf b0[4];
#pragma unroll
    for (int nb = 0; nb < 4; ++nb)
      b0[nb] = load_b16(Wh + (size_t)nb * 16 * D_, D_, kk);
    v16bf a1 = load_a16(Xb, D_, kk + 32);
    v16bf b1[4];
#pragma unroll
    for (int nb = 0; nb < 4; ++nb)
      b1[nb] = load_b16(Wh + (size_t)nb * 16 * D_, D_, kk + 32);

#pragma unroll
    for (int nb = 0; nb < 4; ++nb)
      acc[nb] = wmma_bf16(a0, b0[nb], acc[nb]);
#pragma unroll
    for (int nb = 0; nb < 4; ++nb)
      acc[nb] = wmma_bf16(a1, b1[nb], acc[nb]);
  }

  const int ncol = lane & 15, rowh = (lane >> 4) * 8;
#pragma unroll
  for (int nb = 0; nb < 4; ++nb) {
    float bv_ = bias[h * E_ + nb * 16 + ncol];
    if (which == 2) {
      __bf16* o = vhT + (size_t)bh * E_ * LK_;
#pragma unroll
      for (int i = 0; i < 8; ++i)
        o[(size_t)(nb * 16 + ncol) * LK_ + row0 + i + rowh] = (__bf16)(acc[nb][i] + bv_);
    } else {
      __bf16* o = ((which == 0) ? qh : kh) + ((size_t)bh * LQ_ + row0) * E_;
#pragma unroll
      for (int i = 0; i < 8; ++i)
        o[(size_t)(i + rowh) * E_ + nb * 16 + ncol] = (__bf16)(acc[nb][i] + bv_);
    }
  }
}

// ---------------- flash attention core ----------------
__global__ __launch_bounds__(256)
__attribute__((amdgpu_waves_per_eu(2))) void attn_kernel(
    const __bf16* __restrict__ qh,   // [B*H, LQ, E]
    const __bf16* __restrict__ kh,   // [B*H, LK, E]
    const __bf16* __restrict__ vhT,  // [B*H, E, LK]
    const unsigned char* __restrict__ mask,  // [B, LK] (true = padded)
    __bf16* __restrict__ ctx) {      // [B, LQ, H*E] flat for output GEMM
  const int bh = blockIdx.x, b = bh / H_, h = bh % H_;
  const int wave = threadIdx.x >> 5, lane = threadIdx.x & 31;
  const int q0 = blockIdx.y * 128 + wave * 16;
  const int ncol = lane & 15, rowh = (lane >> 4) * 8;

  __shared__ __bf16 lds[8][16 * 32];   // per-wave P tile (C-layout -> A-layout relay)
  __bf16* pbuf = &lds[wave][0];

  const __bf16* Qb = qh + ((size_t)bh * LQ_ + q0) * E_;
  const __bf16* Kb = kh + (size_t)bh * LK_ * E_;
  const __bf16* VT = vhT + (size_t)bh * E_ * LK_;
  const unsigned char* Mb = mask + (size_t)b * LK_;

  // Q tile is reused for every key chunk: keep both K=E slices in registers.
  v16bf aq0 = load_a16(Qb, E_, 0);
  v16bf aq1 = load_a16(Qb, E_, 32);

  v8f acc[4] = {};
  float mrow[8], lrow[8];
#pragma unroll
  for (int i = 0; i < 8; ++i) { mrow[i] = -__builtin_inff(); lrow[i] = 0.f; }

#pragma unroll 1
  for (int k0 = 0; k0 < LK_; k0 += 32) {
    __builtin_prefetch(Kb + (size_t)(k0 + 32) * E_ + lane * 64, 0, 3);

    // all tiles for this chunk load up front (fresh defs each iteration);
    // previous chunk's ctx WMMAs hide the K-load latency across the back edge,
    // the softmax VALU block below hides the V-load latency.
    v16bf kb0[2], kb1[2];
#pragma unroll
    for (int nb = 0; nb < 2; ++nb) {
      const __bf16* kt = Kb + (size_t)(k0 + nb * 16) * E_;
      kb0[nb] = load_b16(kt, E_, 0);
      kb1[nb] = load_b16(kt, E_, 32);
    }
    v16bf vb_[4];
#pragma unroll
    for (int nb = 0; nb < 4; ++nb)
      vb_[nb] = load_b16(VT + (size_t)nb * 16 * LK_, LK_, k0);

    // scores S[16 x 32] over K=E
    v8f s2[2] = {};
#pragma unroll
    for (int nb = 0; nb < 2; ++nb) {
      s2[nb] = wmma_bf16(aq0, kb0[nb], s2[nb]);
      s2[nb] = wmma_bf16(aq1, kb1[nb], s2[nb]);
    }

    // tau scale + padding mask (column = k0 + nb*16 + ncol, lane-uniform over i)
#pragma unroll
    for (int nb = 0; nb < 2; ++nb) {
      bool pad = Mb[k0 + nb * 16 + ncol] != 0;
#pragma unroll
      for (int i = 0; i < 8; ++i) {
        float sv = s2[nb][i] * TAU;
        s2[nb][i] = pad ? -__builtin_inff() : sv;
      }
    }

    // row max: reduce across the 16 lanes of each half (xor<=8 never mixes halves)
    float mnew[8];
#pragma unroll
    for (int i = 0; i < 8; ++i) {
      float v = fmaxf(s2[0][i], s2[1][i]);
#pragma unroll
      for (int ofs = 1; ofs < 16; ofs <<= 1) v = fmaxf(v, __shfl_xor(v, ofs, 32));
      mnew[i] = fmaxf(mrow[i], v);
    }

    // online-softmax rescale of running state
#pragma unroll
    for (int i = 0; i < 8; ++i) {
      float sc = __expf(mrow[i] - mnew[i]);
      mrow[i] = mnew[i];
      lrow[i] *= sc;
      acc[0][i] *= sc; acc[1][i] *= sc; acc[2][i] *= sc; acc[3][i] *= sc;
    }

    // P = exp(S - m), row-sum, stage to LDS as row-major 16x32 bf16
#pragma unroll
    for (int nb = 0; nb < 2; ++nb) {
#pragma unroll
      for (int i = 0; i < 8; ++i) {
        float pv = __expf(s2[nb][i] - mrow[i]);
        lrow[i] += pv;
        pbuf[(i + rowh) * 32 + nb * 16 + ncol] = (__bf16)pv;
      }
    }
    asm volatile("s_wait_dscnt 0" ::: "memory");  // same-wave DS RAW fence

    // re-load P in A-operand layout; ctx += P(16x32) x V(32x64)
    v16bf pa = load_a16(pbuf, 32, 0);
#pragma unroll
    for (int nb = 0; nb < 4; ++nb)
      acc[nb] = wmma_bf16(pa, vb_[nb], acc[nb]);
  }

  // normalize and emit ctx into [b, q, h*E + e] flat layout (bf16)
  __bf16* o = ctx + ((size_t)b * LQ_ + q0) * HE_ + h * E_;
#pragma unroll
  for (int i = 0; i < 8; ++i) {
    float inv = 1.0f / lrow[i];
#pragma unroll
    for (int nb = 0; nb < 4; ++nb)
      o[(size_t)(i + rowh) * HE_ + nb * 16 + ncol] = (__bf16)(acc[nb][i] * inv);
  }
}

// ---------------- output projection: out = ctx x Wo^T + bo ----------------
__global__ __launch_bounds__(256)
__attribute__((amdgpu_waves_per_eu(2))) void outproj_kernel(
    const __bf16* __restrict__ ctx,  // [B*LQ, HE]
    const __bf16* __restrict__ Wo,   // [HE (n = h*E+e), HE (x, contiguous)]
    const float* __restrict__ bo,    // [HE]
    float* __restrict__ out) {       // [B*LQ, HE] f32
  const int wave = threadIdx.x >> 5, lane = threadIdx.x & 31;
  const int m0 = blockIdx.x * 128 + wave * 16;
  const int n0 = blockIdx.y * 64;
  const __bf16* A = ctx + (size_t)m0 * HE_;
  const __bf16* Wn = Wo + (size_t)n0 * HE_;

  v8f acc[4] = {};
#pragma unroll 1
  for (int kk = 0; kk < HE_; kk += 64) {
    v16bf a0 = load_a16(A, HE_, kk);
    v16bf b0[4];
#pragma unroll
    for (int nb = 0; nb < 4; ++nb)
      b0[nb] = load_b16(Wn + (size_t)nb * 16 * HE_, HE_, kk);
    v16bf a1 = load_a16(A, HE_, kk + 32);
    v16bf b1[4];
#pragma unroll
    for (int nb = 0; nb < 4; ++nb)
      b1[nb] = load_b16(Wn + (size_t)nb * 16 * HE_, HE_, kk + 32);

#pragma unroll
    for (int nb = 0; nb < 4; ++nb)
      acc[nb] = wmma_bf16(a0, b0[nb], acc[nb]);
#pragma unroll
    for (int nb = 0; nb < 4; ++nb)
      acc[nb] = wmma_bf16(a1, b1[nb], acc[nb]);
  }

  const int ncol = lane & 15, rowh = (lane >> 4) * 8;
#pragma unroll
  for (int nb = 0; nb < 4; ++nb) {
    float bv = bo[n0 + nb * 16 + ncol];
#pragma unroll
    for (int i = 0; i < 8; ++i)
      out[(size_t)(m0 + i + rowh) * HE_ + n0 + nb * 16 + ncol] = acc[nb][i] + bv;
  }
}

extern "C" void kernel_launch(void* const* d_in, const int* in_sizes, int n_in,
                              void* d_out, int out_size, void* d_ws, size_t ws_size,
                              hipStream_t stream) {
  (void)in_sizes; (void)n_in; (void)out_size; (void)ws_size;
  const float* q  = (const float*)d_in[0];
  const float* k  = (const float*)d_in[1];
  const float* v  = (const float*)d_in[2];
  const unsigned char* mask = (const unsigned char*)d_in[3];
  const float* Wq = (const float*)d_in[4];
  const float* bq = (const float*)d_in[5];
  const float* Wk = (const float*)d_in[6];
  const float* bk = (const float*)d_in[7];
  const float* Wv = (const float*)d_in[8];
  const float* bv = (const float*)d_in[9];
  const float* Wo = (const float*)d_in[10];
  const float* bo = (const float*)d_in[11];
  float* out = (float*)d_out;

  // workspace carve-up (bf16), total ~64 MB
  __bf16* p = (__bf16*)d_ws;
  const size_t nQKV = (size_t)B_ * LQ_ * D_;       // 4M each
  const size_t nW   = (size_t)H_ * E_ * D_;        // 1M each
  const size_t nH   = (size_t)B_ * H_ * LQ_ * E_;  // 4M each
  __bf16* q_bf = p; p += nQKV;
  __bf16* k_bf = p; p += nQKV;
  __bf16* v_bf = p; p += nQKV;
  __bf16* Wq_bf = p; p += nW;
  __bf16* Wk_bf = p; p += nW;
  __bf16* Wv_bf = p; p += nW;
  __bf16* Wo_bf = p; p += nW;
  __bf16* qh  = p; p += nH;
  __bf16* kh  = p; p += nH;
  __bf16* vhT = p; p += nH;
  __bf16* ctx = p; p += nQKV;

  dim3 blk(256);
  auto cvt_blocks = [](size_t n) { return dim3((unsigned)((n / 4 + 255) / 256)); };
  cvt_f32_bf16<<<cvt_blocks(nQKV), blk, 0, stream>>>(q, q_bf, (int)nQKV);
  cvt_f32_bf16<<<cvt_blocks(nQKV), blk, 0, stream>>>(k, k_bf, (int)nQKV);
  cvt_f32_bf16<<<cvt_blocks(nQKV), blk, 0, stream>>>(v, v_bf, (int)nQKV);
  cvt_f32_bf16<<<cvt_blocks(nW),   blk, 0, stream>>>(Wq, Wq_bf, (int)nW);
  cvt_f32_bf16<<<cvt_blocks(nW),   blk, 0, stream>>>(Wk, Wk_bf, (int)nW);
  cvt_f32_bf16<<<cvt_blocks(nW),   blk, 0, stream>>>(Wv, Wv_bf, (int)nW);
  cvt_f32_bf16<<<cvt_blocks(nW),   blk, 0, stream>>>(Wo, Wo_bf, (int)nW);

  proj_kernel<<<dim3(B_ * H_, LQ_ / 128, 3), blk, 0, stream>>>(
      q_bf, k_bf, v_bf, Wq_bf, Wk_bf, Wv_bf, bq, bk, bv, qh, kh, vhT);

  attn_kernel<<<dim3(B_ * H_, LQ_ / 128), blk, 0, stream>>>(qh, kh, vhT, mask, ctx);

  outproj_kernel<<<dim3((B_ * LQ_) / 128, HE_ / 64), blk, 0, stream>>>(
      ctx, Wo_bf, bo, out);
}
